// DecoderLayer_14053132992618
// MI455X (gfx1250) — compile-verified, matrix operand order checked by reading, not run on hardware
//
#include <hip/hip_runtime.h>
#include <cstdint>
#include <cstddef>

// ---------------------------------------------------------------------------
// Types
// ---------------------------------------------------------------------------
typedef __bf16 bf16;
typedef __attribute__((ext_vector_type(16))) __bf16 v16bf;
typedef __attribute__((ext_vector_type(8)))  __bf16 bf16x8;
typedef __attribute__((ext_vector_type(8)))  float  v8f;
typedef __attribute__((ext_vector_type(4)))  float  f32x4;

static constexpr int B_  = 4;
static constexpr int T_  = 1024;
static constexpr int S_  = 1024;
static constexpr int D_  = 1024;
static constexpr int H_  = 16;
static constexpr int DK_ = 64;      // D / H
static constexpr int F_  = 4096;
static constexpr int BT_ = B_ * T_; // 4096 rows of the activation matrix

__device__ __forceinline__ v8f zero8() {
    v8f z = {0.f, 0.f, 0.f, 0.f, 0.f, 0.f, 0.f, 0.f};
    return z;
}

// D = A(16x32 bf16) * B(32x16 bf16) + C(16x16 f32)
__device__ __forceinline__ v8f wmma_bf16(v16bf a, v16bf b, v8f c) {
    return __builtin_amdgcn_wmma_f32_16x16x32_bf16(
        /*neg_a=*/false, a, /*neg_b=*/false, b,
        /*c_mod=*/(short)0, c, /*reuse_a=*/false, /*reuse_b=*/false);
}

// Load a 16x32 bf16 fragment (A-layout; B-layout is identical with "row"=N and
// the K axis contiguous).  base points at element (row0, k0) of the tile.
//   lanes 0-15 : row = lane,     K = {0..7, 16..23}
//   lanes16-31 : row = lane-16,  K = {8..15, 24..31}
__device__ __forceinline__ v16bf load_frag(const bf16* __restrict__ base, int stride) {
    const int lane = threadIdx.x & 31;
    const int lo = lane & 15, hi = lane >> 4;
    const bf16* p = base + (size_t)lo * stride + hi * 8;
    bf16x8 a = *(const bf16x8*)(p);
    bf16x8 c = *(const bf16x8*)(p + 16);
    v16bf f;
#pragma unroll
    for (int i = 0; i < 8; ++i) { f[i] = a[i]; f[i + 8] = c[i]; }
    return f;
}

// ---------------------------------------------------------------------------
// CDNA5 async global->LDS (ASYNCcnt-tracked DMA path, bypasses VGPRs).
// VDST = 32-bit LDS byte address (low 32 bits of the generic shared pointer,
// per the ISA flat-aperture rule).
// ---------------------------------------------------------------------------
__device__ __forceinline__ void async_copy_32B(unsigned lds, unsigned long long g) {
    asm volatile("global_load_async_to_lds_b128 %0, %1, off"
                 :: "v"(lds), "v"(g) : "memory");
    asm volatile("global_load_async_to_lds_b128 %0, %1, off"
                 :: "v"(lds + 16u), "v"(g + 16ull) : "memory");
}
__device__ __forceinline__ void async_wait0() {
    asm volatile("s_wait_asynccnt 0x0" ::: "memory");
}

// ---------------------------------------------------------------------------
// Elementwise cast f32 -> bf16 (also used to pre-convert all weights once)
// ---------------------------------------------------------------------------
__global__ void cast_f32_bf16(const float* __restrict__ x, bf16* __restrict__ y, size_t n) {
    size_t i = (size_t)blockIdx.x * blockDim.x + threadIdx.x;
    size_t stride = (size_t)gridDim.x * blockDim.x;
    for (; i < n; i += stride) y[i] = (bf16)x[i];
}

// ---------------------------------------------------------------------------
// LayerNorm over last dim (Dm = 1024), one row per 256-thread block, bf16 out
// ---------------------------------------------------------------------------
__global__ __launch_bounds__(256) void layernorm_bf16(
    const float* __restrict__ x, const float* __restrict__ g,
    const float* __restrict__ be, bf16* __restrict__ y, int Dm) {
    __shared__ float red[2][8];
    const int row  = blockIdx.x;
    const int lane = threadIdx.x & 31, w = threadIdx.x >> 5;
    const int c0   = threadIdx.x * 4;
    const float* xr = x + (size_t)row * Dm;
    f32x4 xv = *(const f32x4*)(xr + c0);
    float s  = xv[0] + xv[1] + xv[2] + xv[3];
    float s2 = xv[0]*xv[0] + xv[1]*xv[1] + xv[2]*xv[2] + xv[3]*xv[3];
#pragma unroll
    for (int d = 1; d < 32; d <<= 1) {
        s  += __shfl_xor(s,  d, 32);
        s2 += __shfl_xor(s2, d, 32);
    }
    if (lane == 0) { red[0][w] = s; red[1][w] = s2; }
    __syncthreads();
    s = 0.f; s2 = 0.f;
#pragma unroll
    for (int i = 0; i < 8; ++i) { s += red[0][i]; s2 += red[1][i]; }
    const float mean = s / Dm;
    const float var  = s2 / Dm - mean * mean;
    const float r    = rsqrtf(var + 1e-5f);
    bf16* yr = y + (size_t)row * Dm;
#pragma unroll
    for (int c = 0; c < 4; ++c)
        yr[c0 + c] = (bf16)((xv[c] - mean) * r * g[c0 + c] + be[c0 + c]);
}

// ---------------------------------------------------------------------------
// bf16 transpose: v[b][s][d] -> vt[b][d][s]
// ---------------------------------------------------------------------------
__global__ __launch_bounds__(256) void transpose_bf16(
    const bf16* __restrict__ v, bf16* __restrict__ vt, int S, int Dm) {
    __shared__ bf16 t[32][33];
    const int b  = blockIdx.z;
    const int s0 = blockIdx.x * 32, d0 = blockIdx.y * 32;
    const int tx = threadIdx.x & 31, ty = threadIdx.x >> 5;   // 8 row-groups
#pragma unroll
    for (int r = ty; r < 32; r += 8)
        t[r][tx] = v[((size_t)b * S + s0 + r) * Dm + d0 + tx];
    __syncthreads();
#pragma unroll
    for (int r = ty; r < 32; r += 8)
        vt[((size_t)b * Dm + d0 + r) * S + s0 + tx] = t[tx][r];
}

// ---------------------------------------------------------------------------
// WMMA GEMM:  C[M][N] = act( A[M][K](bf16) * W[N][K](bf16)^T + bias[N] )
//             (+ optional f32 residual), write f32 and/or bf16.
// Block tile 128x128, 8 waves (2x4), wave tile 64x32, K-step 32.
// Double-buffered LDS filled by async global->LDS DMA (ASYNCcnt).
// ---------------------------------------------------------------------------
__global__ __launch_bounds__(256) void gemm_wmma(
    const bf16*  __restrict__ A, const bf16* __restrict__ W,
    const float* __restrict__ bias, const float* __restrict__ resid,
    float* __restrict__ outF, bf16* __restrict__ outB,
    int M, int N, int K, int relu) {
    constexpr int LDSW = 40;                  // padded row stride (bank-spread)
    __shared__ bf16 As[2][128 * LDSW];
    __shared__ bf16 Ws[2][128 * LDSW];
    const int m0 = blockIdx.y * 128, n0 = blockIdx.x * 128;
    const int tid  = threadIdx.x;
    const int lane = tid & 31, w = tid >> 5;
    const int mw = (w >> 2) * 64;             // 0 / 64
    const int nw = (w & 3) * 32;              // 0..96
    const int arow = tid >> 1, aseg = (tid & 1) * 16;

    // Per-thread source rows and LDS destinations (32B per thread per tile)
    const bf16* gA = A + (size_t)(m0 + arow) * K + aseg;
    const bf16* gW = W + (size_t)(n0 + arow) * K + aseg;
    unsigned ldsA[2], ldsW[2];
#pragma unroll
    for (int bfi = 0; bfi < 2; ++bfi) {
        ldsA[bfi] = (unsigned)(uintptr_t)&As[bfi][arow * LDSW + aseg];
        ldsW[bfi] = (unsigned)(uintptr_t)&Ws[bfi][arow * LDSW + aseg];
    }

    v8f acc[4][2];
#pragma unroll
    for (int i = 0; i < 4; ++i)
#pragma unroll
        for (int j = 0; j < 2; ++j) acc[i][j] = zero8();

    // Preload tile 0 via async DMA
    async_copy_32B(ldsA[0], (unsigned long long)(uintptr_t)gA);
    async_copy_32B(ldsW[0], (unsigned long long)(uintptr_t)gW);
    async_wait0();
    __syncthreads();

    int buf = 0;
    for (int kt = 0; kt < K; kt += 32) {
        // Kick off next tile into the other buffer (overlaps with WMMA below)
        const int nxt = kt + 32;
        if (nxt < K) {
            async_copy_32B(ldsA[buf ^ 1], (unsigned long long)(uintptr_t)(gA + nxt));
            async_copy_32B(ldsW[buf ^ 1], (unsigned long long)(uintptr_t)(gW + nxt));
        }

        // Compute on current buffer
        v16bf af[4], bfr[2];
#pragma unroll
        for (int i = 0; i < 4; ++i)
            af[i]  = load_frag(&As[buf][(mw + i * 16) * LDSW], LDSW);
#pragma unroll
        for (int j = 0; j < 2; ++j)
            bfr[j] = load_frag(&Ws[buf][(nw + j * 16) * LDSW], LDSW);
#pragma unroll
        for (int i = 0; i < 4; ++i)
#pragma unroll
            for (int j = 0; j < 2; ++j)
                acc[i][j] = wmma_bf16(af[i], bfr[j], acc[i][j]);

        if (nxt < K) async_wait0();
        __syncthreads();
        buf ^= 1;
    }

    // Epilogue (C-layout: VGPR g -> row g (lanes 0-15) / row 8+g (lanes 16-31))
    const int lo = lane & 15, hi = lane >> 4;
#pragma unroll
    for (int i = 0; i < 4; ++i)
#pragma unroll
        for (int j = 0; j < 2; ++j)
#pragma unroll
            for (int g = 0; g < 8; ++g) {
                const int row = m0 + mw + i * 16 + (hi ? 8 + g : g);
                const int col = n0 + nw + j * 16 + lo;
                float v = acc[i][j][g] + bias[col];
                if (relu) v = fmaxf(v, 0.f);
                const size_t idx = (size_t)row * N + col;
                if (resid) v += resid[idx];
                if (outF) outF[idx] = v;
                if (outB) outB[idx] = (bf16)v;
            }
}

// ---------------------------------------------------------------------------
// Flash attention: one wave per (b, h, 16 Q rows).  dk = 64.
// Q/K : [B*rows, D] bf16 (head slice = cols h*64..h*64+63)
// VT  : [B, D, Skv]  bf16 (pre-transposed V)
// O   : [B*Tq, D]    bf16
// ---------------------------------------------------------------------------
__global__ __launch_bounds__(32) void attn_wmma(
    const bf16* __restrict__ Q, const bf16* __restrict__ Km,
    const bf16* __restrict__ VT, bf16* __restrict__ O,
    const int* __restrict__ mask, const float* __restrict__ sbias,
    const float* __restrict__ scale_p, int Tq, int Skv, int causal) {
    constexpr int LDSW = 40;
    __shared__ bf16 pls[16 * LDSW];
    const int lane = threadIdx.x & 31;
    const int lo = lane & 15, hi = lane >> 4;
    const int bh = blockIdx.y;
    const int b = bh / H_, h = bh % H_;
    const int q0 = blockIdx.x * 16;
    const float sscale = scale_p ? *scale_p : 0.f;

    // Resident Q fragments (16 x 64 = two K-steps of 32)
    v16bf qf[2];
#pragma unroll
    for (int kf = 0; kf < 2; ++kf)
        qf[kf] = load_frag(Q + ((size_t)(b * Tq + q0)) * D_ + h * DK_ + kf * 32, D_);

    float mrun[8], lrun[8];
#pragma unroll
    for (int g = 0; g < 8; ++g) { mrun[g] = -1e30f; lrun[g] = 0.f; }
    v8f o[4];
#pragma unroll
    for (int f = 0; f < 4; ++f) o[f] = zero8();

    const int send = causal ? min(Skv, q0 + 16) : Skv;
    for (int s0 = 0; s0 < send; s0 += 32) {
        // scores: 16 x 32 (two C fragments)
        v8f sc[2] = {zero8(), zero8()};
#pragma unroll
        for (int j = 0; j < 2; ++j)
#pragma unroll
            for (int kf = 0; kf < 2; ++kf) {
                v16bf kb = load_frag(
                    Km + ((size_t)(b * Skv + s0 + j * 16)) * D_ + h * DK_ + kf * 32, D_);
                sc[j] = wmma_bf16(qf[kf], kb, sc[j]);
            }
        // scale (1/sqrt(64)), sentence bias, padding mask, causal mask
#pragma unroll
        for (int j = 0; j < 2; ++j) {
            const int col = s0 + j * 16 + lo;
            const float addb = sbias ? sscale * sbias[b * Skv + col] : 0.f;
            const bool mz = mask ? (mask[b * Skv + col] == 0) : false;
#pragma unroll
            for (int g = 0; g < 8; ++g) {
                float v = sc[j][g] * 0.125f + addb;
                if (mz) v = -1e9f;
                if (causal) {
                    const int row = q0 + (hi ? 8 + g : g);
                    if (col > row) v = -1e9f;
                }
                sc[j][g] = v;
            }
        }
        // Online softmax: 16-lane row reductions (rows align with o[] layout)
        float corr[8];
#pragma unroll
        for (int g = 0; g < 8; ++g) {
            float mx = fmaxf(sc[0][g], sc[1][g]);
#pragma unroll
            for (int d = 1; d < 16; d <<= 1) mx = fmaxf(mx, __shfl_xor(mx, d, 32));
            const float mnew = fmaxf(mrun[g], mx);
            const float p0 = expf(sc[0][g] - mnew);
            const float p1 = expf(sc[1][g] - mnew);
            float rs = p0 + p1;
#pragma unroll
            for (int d = 1; d < 16; d <<= 1) rs += __shfl_xor(rs, d, 32);
            corr[g] = expf(mrun[g] - mnew);
            lrun[g] = lrun[g] * corr[g] + rs;
            mrun[g] = mnew;
            sc[0][g] = p0; sc[1][g] = p1;
        }
#pragma unroll
        for (int f = 0; f < 4; ++f)
#pragma unroll
            for (int g = 0; g < 8; ++g) o[f][g] *= corr[g];

        // C-layout P -> LDS (row major 16x32) -> A-layout fragment
        __syncthreads();
#pragma unroll
        for (int g = 0; g < 8; ++g) {
            const int r = hi ? 8 + g : g;
            pls[r * LDSW + lo]      = (bf16)sc[0][g];
            pls[r * LDSW + 16 + lo] = (bf16)sc[1][g];
        }
        __syncthreads();
        const v16bf pf = load_frag(&pls[0], LDSW);

        // O += P(16x32) * V(32x64), V columns contiguous in VT
#pragma unroll
        for (int nf = 0; nf < 4; ++nf) {
            v16bf vb = load_frag(
                VT + ((size_t)b * D_ + h * DK_ + nf * 16) * Skv + s0, Skv);
            o[nf] = wmma_bf16(pf, vb, o[nf]);
        }
    }

    // Finalize and store
#pragma unroll
    for (int nf = 0; nf < 4; ++nf)
#pragma unroll
        for (int g = 0; g < 8; ++g) {
            const int row = q0 + (hi ? 8 + g : g);
            const int col = h * DK_ + nf * 16 + lo;
            O[((size_t)(b * Tq + row)) * D_ + col] = (bf16)(o[nf][g] / lrun[g]);
        }
}

// ---------------------------------------------------------------------------
// Host orchestration
// ---------------------------------------------------------------------------
extern "C" void kernel_launch(void* const* d_in, const int* in_sizes, int n_in,
                              void* d_out, int out_size, void* d_ws, size_t ws_size,
                              hipStream_t stream) {
    (void)in_sizes; (void)n_in; (void)out_size; (void)ws_size;

    const float* x      = (const float*)d_in[0];
    const float* memory = (const float*)d_in[1];
    const float* sbias  = (const float*)d_in[2];
    const int*   trgm   = (const int*)d_in[3];
    // d_in[4] = causal mask (recomputed analytically on device)
    const int*   srcm   = (const int*)d_in[5];
    const float* sa_wq = (const float*)d_in[6],  *sa_bq = (const float*)d_in[7];
    const float* sa_wk = (const float*)d_in[8],  *sa_bk = (const float*)d_in[9];
    const float* sa_wv = (const float*)d_in[10], *sa_bv = (const float*)d_in[11];
    const float* sa_wo = (const float*)d_in[12], *sa_bo = (const float*)d_in[13];
    // d_in[14] = sa_scale (unused: self-attention has no sentence bias)
    const float* ca_wq = (const float*)d_in[15], *ca_bq = (const float*)d_in[16];
    const float* ca_wk = (const float*)d_in[17], *ca_bk = (const float*)d_in[18];
    const float* ca_wv = (const float*)d_in[19], *ca_bv = (const float*)d_in[20];
    const float* ca_wo = (const float*)d_in[21], *ca_bo = (const float*)d_in[22];
    const float* ca_scale = (const float*)d_in[23];
    const float* ln1_g = (const float*)d_in[24], *ln1_b = (const float*)d_in[25];
    const float* ln2_g = (const float*)d_in[26], *ln2_b = (const float*)d_in[27];
    const float* ln3_g = (const float*)d_in[28], *ln3_b = (const float*)d_in[29];
    const float* ffn_w1 = (const float*)d_in[30], *ffn_b1 = (const float*)d_in[31];
    const float* ffn_w2 = (const float*)d_in[32], *ffn_b2 = (const float*)d_in[33];

    // Workspace carve-up
    const size_t BTD = (size_t)BT_ * D_;
    const size_t DD  = (size_t)D_ * D_;
    const size_t FD  = (size_t)F_ * D_;
    char* ws = (char*)d_ws;
    size_t off = 0;
    auto take = [&](size_t bytes) { char* p = ws + off; off = (off + bytes + 255) & ~(size_t)255; return p; };
    bf16* x0b   = (bf16*)take(BTD * 2);           // bf16 of residual (SA K/V src)
    bf16* normb = (bf16*)take(BTD * 2);           // LN output
    bf16* qb    = (bf16*)take(BTD * 2);
    bf16* kb    = (bf16*)take(BTD * 2);
    bf16* vb    = (bf16*)take(BTD * 2);
    bf16* vtb   = (bf16*)take(BTD * 2);           // V transposed [B, D, S]
    bf16* attnb = (bf16*)take(BTD * 2);
    bf16* memb  = (bf16*)take((size_t)B_ * S_ * D_ * 2);
    bf16* ffnh  = (bf16*)take((size_t)BT_ * F_ * 2);
    float* res1 = (float*)take(BTD * 4);
    float* res2 = (float*)take(BTD * 4);
    // bf16 weight copies (converted once; L2-resident thereafter)
    bf16* wb[10];
    for (int i = 0; i < 8; ++i) wb[i] = (bf16*)take(DD * 2);
    wb[8] = (bf16*)take(FD * 2);
    wb[9] = (bf16*)take(FD * 2);

    // Pre-convert all weights f32 -> bf16
    const float* wsrc[10] = {sa_wq, sa_wk, sa_wv, sa_wo, ca_wq, ca_wk, ca_wv, ca_wo,
                             ffn_w1, ffn_w2};
    for (int i = 0; i < 10; ++i) {
        const size_t n = (i < 8) ? DD : FD;
        cast_f32_bf16<<<1024, 256, 0, stream>>>(wsrc[i], wb[i], n);
    }

    const dim3 gemmDD(D_ / 128, BT_ / 128);       // (8, 32)
    const dim3 gemmDF(F_ / 128, BT_ / 128);       // (32, 32)
    const dim3 attnG(T_ / 16, B_ * H_);           // (64, 64)
    const dim3 trG(S_ / 32, D_ / 32, B_);         // (32, 32, 4)

    // ---- Self-attention block (Q from LN1(x), K/V from raw x) ----
    cast_f32_bf16<<<4096, 256, 0, stream>>>(x, x0b, BTD);
    layernorm_bf16<<<BT_, 256, 0, stream>>>(x, ln1_g, ln1_b, normb, D_);
    gemm_wmma<<<gemmDD, 256, 0, stream>>>(normb, wb[0], sa_bq, nullptr, nullptr, qb, BT_, D_, D_, 0);
    gemm_wmma<<<gemmDD, 256, 0, stream>>>(x0b,   wb[1], sa_bk, nullptr, nullptr, kb, BT_, D_, D_, 0);
    gemm_wmma<<<gemmDD, 256, 0, stream>>>(x0b,   wb[2], sa_bv, nullptr, nullptr, vb, BT_, D_, D_, 0);
    transpose_bf16<<<trG, 256, 0, stream>>>(vb, vtb, T_, D_);
    attn_wmma<<<attnG, 32, 0, stream>>>(qb, kb, vtb, attnb, trgm, nullptr, nullptr, T_, T_, /*causal=*/1);
    gemm_wmma<<<gemmDD, 256, 0, stream>>>(attnb, wb[3], sa_bo, x, res1, nullptr, BT_, D_, D_, 0);

    // ---- Cross-attention block (K/V from memory, sentence bias + src mask) ----
    cast_f32_bf16<<<4096, 256, 0, stream>>>(memory, memb, (size_t)B_ * S_ * D_);
    layernorm_bf16<<<BT_, 256, 0, stream>>>(res1, ln2_g, ln2_b, normb, D_);
    gemm_wmma<<<gemmDD, 256, 0, stream>>>(normb, wb[4], ca_bq, nullptr, nullptr, qb, BT_, D_, D_, 0);
    gemm_wmma<<<gemmDD, 256, 0, stream>>>(memb,  wb[5], ca_bk, nullptr, nullptr, kb, B_ * S_, D_, D_, 0);
    gemm_wmma<<<gemmDD, 256, 0, stream>>>(memb,  wb[6], ca_bv, nullptr, nullptr, vb, B_ * S_, D_, D_, 0);
    transpose_bf16<<<trG, 256, 0, stream>>>(vb, vtb, S_, D_);
    attn_wmma<<<attnG, 32, 0, stream>>>(qb, kb, vtb, attnb, srcm, sbias, ca_scale, T_, S_, /*causal=*/0);
    gemm_wmma<<<gemmDD, 256, 0, stream>>>(attnb, wb[7], ca_bo, res1, res2, nullptr, BT_, D_, D_, 0);

    // ---- FFN block ----
    layernorm_bf16<<<BT_, 256, 0, stream>>>(res2, ln3_g, ln3_b, normb, D_);
    gemm_wmma<<<gemmDF, 256, 0, stream>>>(normb, wb[8], ffn_b1, nullptr, nullptr, ffnh, BT_, F_, D_, /*relu=*/1);
    gemm_wmma<<<gemmDD, 256, 0, stream>>>(ffnh, wb[9], ffn_b2, res2, (float*)d_out, nullptr, BT_, D_, F_, 0);
}
